// JointTransformerBlockControl_17755394801770
// MI455X (gfx1250) — matrix operand order repaired and checked by reading, not compiled, not run on hardware
//
#include <hip/hip_runtime.h>

#define DIM   1536
#define IMG   1024
#define TXT   256
#define SEQ   1280
#define NHEAD 24

typedef __attribute__((ext_vector_type(16))) __bf16       v16bf;
typedef __attribute__((ext_vector_type(8)))  float        v8f;
typedef __attribute__((ext_vector_type(8)))  unsigned int v8u;

union FragBF { v16bf bf; v8u u; };

__device__ inline __bf16 f2bf(float x) { return (__bf16)x; }

__device__ inline unsigned pack_bf16(float lo, float hi) {
  unsigned short a = __builtin_bit_cast(unsigned short, (__bf16)lo);
  unsigned short b = __builtin_bit_cast(unsigned short, (__bf16)hi);
  return (unsigned)a | ((unsigned)b << 16);
}
__device__ inline float bf16_lo(unsigned u) {
  return (float)__builtin_bit_cast(__bf16, (unsigned short)(u & 0xffffu));
}
__device__ inline float bf16_hi(unsigned u) {
  return (float)__builtin_bit_cast(__bf16, (unsigned short)(u >> 16));
}

// ---- WMMA fragment loaders (LDS -> VGPR), per CDNA5 ISA 16-bit layouts ----
// A fragment: 16x32 bf16 (MxK). Element (m,k) at buf[m*ldk + k]; even-k pairs contiguous.
__device__ inline FragBF load_a(const __bf16* buf, int ldk) {
  int lane = threadIdx.x & 31;
  int row = lane & 15, hi = lane >> 4;
  FragBF f;
#pragma unroll
  for (int j = 0; j < 8; ++j) {
    int k = (j < 4) ? (8 * hi + 2 * j) : (16 + 8 * hi + 2 * (j - 4));
    f.u[j] = *(const unsigned int*)(buf + row * ldk + k);
  }
  return f;
}
// B fragment: 32x16 bf16 (KxN). Element (k,n) at buf[n*ldn + k]; pairs contiguous in k.
__device__ inline FragBF load_b(const __bf16* buf, int ldn) {
  int lane = threadIdx.x & 31;
  int col = lane & 15, hi = lane >> 4;
  FragBF f;
#pragma unroll
  for (int j = 0; j < 8; ++j) {
    int k = (j < 4) ? (8 * hi + 2 * j) : (16 + 8 * hi + 2 * (j - 4));
    f.u[j] = *(const unsigned int*)(buf + col * ldn + k);
  }
  return f;
}
__device__ inline v8f wmma_bf16(FragBF a, FragBF b, v8f c) {
  return __builtin_amdgcn_wmma_f32_16x16x32_bf16(false, a.bf, false, b.bf,
                                                 (short)0, c, false, false);
}

// ---- gfx1250 async global->LDS copy (ASYNCcnt-tracked, ISA 10.7/15.18) ----
__device__ inline unsigned lds_off32(const void* p) {
  // generic LDS address: addr[31:0] is the group-segment byte offset
  return (unsigned)(unsigned long long)p;
}
__device__ inline void async_copy_b128(unsigned ldsoff, const void* gaddr) {
  asm volatile("global_load_async_to_lds_b128 %0, %1, off"
               :: "v"(ldsoff), "v"(gaddr) : "memory");
}
__device__ inline void wait_async0() {
  asm volatile("s_wait_asynccnt 0x0" ::: "memory");
}

// ---- double-buffered GEMM core: acc(2x2 WMMA tiles/wave) over K=1536 ----
__device__ inline void gemm_tiles(const float* __restrict__ A,
                                  const float* __restrict__ W,
                                  int m0, int n0, int wm, int wn,
                                  __bf16* sA, __bf16* sBt, v8f acc[2][2])
{
  const int t = threadIdx.x;
  float4 ra[4], rw[2];

  auto load_regs = [&](int k0) {
#pragma unroll
    for (int it = 0; it < 4; ++it) {         // A tile 128x32: 1024 float4
      int i = t + it * 256;
      int row = i >> 3, c4 = (i & 7) << 2;
      ra[it] = *(const float4*)(A + (size_t)(m0 + row) * DIM + k0 + c4);
    }
#pragma unroll
    for (int it = 0; it < 2; ++it) {         // W tile 32x64: 512 float4
      int i = t + it * 256;
      int kr = i >> 4, c4 = (i & 15) << 2;
      rw[it] = *(const float4*)(W + (size_t)(k0 + kr) * DIM + n0 + c4);
    }
  };
  auto store_lds = [&](int buf) {
    __bf16* dA = sA + buf * (128 * 32);
    __bf16* dB = sBt + buf * (64 * 32);
#pragma unroll
    for (int it = 0; it < 4; ++it) {
      int i = t + it * 256;
      int row = i >> 3, c4 = (i & 7) << 2;
      __bf16* d = dA + row * 32 + c4;
      d[0] = f2bf(ra[it].x); d[1] = f2bf(ra[it].y);
      d[2] = f2bf(ra[it].z); d[3] = f2bf(ra[it].w);
    }
#pragma unroll
    for (int it = 0; it < 2; ++it) {         // transpose W -> [n][k]
      int i = t + it * 256;
      int kr = i >> 4, c4 = (i & 15) << 2;
      dB[(c4 + 0) * 32 + kr] = f2bf(rw[it].x);
      dB[(c4 + 1) * 32 + kr] = f2bf(rw[it].y);
      dB[(c4 + 2) * 32 + kr] = f2bf(rw[it].z);
      dB[(c4 + 3) * 32 + kr] = f2bf(rw[it].w);
    }
  };

#pragma unroll
  for (int i = 0; i < 2; ++i)
#pragma unroll
    for (int j = 0; j < 2; ++j)
#pragma unroll
      for (int r = 0; r < 8; ++r) acc[i][j][r] = 0.f;

  load_regs(0);
  store_lds(0);
  __syncthreads();
  int cur = 0;
  for (int k0 = 0; k0 < DIM; k0 += 32) {
    bool hn = (k0 + 32) < DIM;
    if (hn) load_regs(k0 + 32);              // global loads overlap WMMAs below
    const __bf16* cA = sA + cur * (128 * 32);
    const __bf16* cB = sBt + cur * (64 * 32);
    FragBF a0 = load_a(cA + (wm + 0) * 32, 32);
    FragBF a1 = load_a(cA + (wm + 16) * 32, 32);
    FragBF b0 = load_b(cB + (wn + 0) * 32, 32);
    FragBF b1 = load_b(cB + (wn + 16) * 32, 32);
    acc[0][0] = wmma_bf16(a0, b0, acc[0][0]);
    acc[0][1] = wmma_bf16(a0, b1, acc[0][1]);
    acc[1][0] = wmma_bf16(a1, b0, acc[1][0]);
    acc[1][1] = wmma_bf16(a1, b1, acc[1][1]);
    if (hn) store_lds(cur ^ 1);
    __syncthreads();
    cur ^= 1;
  }
}

// C f32 out (V projections, output projections)
__global__ void __launch_bounds__(256)
gemm_bias_f32_kernel(const float* __restrict__ A, const float* __restrict__ W,
                     const float* __restrict__ bias, float* __restrict__ C, int M)
{
  __shared__ alignas(16) __bf16 sA[2 * 128 * 32];
  __shared__ alignas(16) __bf16 sBt[2 * 64 * 32];
  const int lane = threadIdx.x & 31, w = threadIdx.x >> 5;
  const int wm = (w & 3) * 32, wn = (w >> 2) * 32;
  const int m0 = blockIdx.y * 128, n0 = blockIdx.x * 64;
  v8f acc[2][2];
  gemm_tiles(A, W, m0, n0, wm, wn, sA, sBt, acc);
#pragma unroll
  for (int mi = 0; mi < 2; ++mi)
#pragma unroll
    for (int ni = 0; ni < 2; ++ni)
#pragma unroll
      for (int r = 0; r < 8; ++r) {
        int m = m0 + wm + mi * 16 + r + 8 * (lane >> 4);
        int n = n0 + wn + ni * 16 + (lane & 15);
        if (m < M) C[(size_t)m * DIM + n] = acc[mi][ni][r] + bias[n];
      }
}

// C bf16 out (Q/K projections feeding attention)
__global__ void __launch_bounds__(256)
gemm_bias_bf16_kernel(const float* __restrict__ A, const float* __restrict__ W,
                      const float* __restrict__ bias, __bf16* __restrict__ C, int M)
{
  __shared__ alignas(16) __bf16 sA[2 * 128 * 32];
  __shared__ alignas(16) __bf16 sBt[2 * 64 * 32];
  const int lane = threadIdx.x & 31, w = threadIdx.x >> 5;
  const int wm = (w & 3) * 32, wn = (w >> 2) * 32;
  const int m0 = blockIdx.y * 128, n0 = blockIdx.x * 64;
  v8f acc[2][2];
  gemm_tiles(A, W, m0, n0, wm, wn, sA, sBt, acc);
#pragma unroll
  for (int mi = 0; mi < 2; ++mi)
#pragma unroll
    for (int ni = 0; ni < 2; ++ni)
#pragma unroll
      for (int r = 0; r < 8; ++r) {
        int m = m0 + wm + mi * 16 + r + 8 * (lane >> 4);
        int n = n0 + wn + ni * 16 + (lane & 15);
        if (m < M) C[(size_t)m * DIM + n] = f2bf(acc[mi][ni][r] + bias[n]);
      }
}

// Flash attention: Q,K bf16 (raw async-to-LDS copies, double buffered),
// V f32 (batched loads + transposed bf16 staging), O f32.
// One head x 64 q-rows per block, 4 waves, online softmax over 32-key tiles.
__global__ void __launch_bounds__(128)
flash_attn_kernel(const __bf16* __restrict__ Q, const __bf16* __restrict__ K,
                  const float* __restrict__ V, float* __restrict__ O, int Sk)
{
  __shared__ alignas(16) __bf16 sQ[64 * 64];
  __shared__ alignas(16) __bf16 sK[2][32 * 64];    // [key][d]
  __shared__ alignas(16) __bf16 sVt[2][64 * 32];   // [d][key]
  __shared__ alignas(16) __bf16 sP[4][16 * 32];    // per-wave P bounce
  const int t = threadIdx.x;
  const int lane = t & 31, w = t >> 5;
  const int hbase = blockIdx.y * 64;
  const int q0 = blockIdx.x * 64;
  const float scale = 0.125f;  // 1/sqrt(64), applied to S post-WMMA

  // async copy Q: 64 rows x 128B
#pragma unroll
  for (int it = 0; it < 4; ++it) {
    int i = t + it * 128;
    int row = i >> 3, ch = i & 7;
    async_copy_b128(lds_off32(sQ) + row * 128 + ch * 16,
                    Q + (size_t)(q0 + row) * DIM + hbase + ch * 8);
  }
  auto asyncK = [&](int kt, int buf) {           // 32 rows x 128B
#pragma unroll
    for (int it = 0; it < 2; ++it) {
      int i = t + it * 128;
      int key = i >> 3, ch = i & 7;
      async_copy_b128(lds_off32(sK[buf]) + key * 128 + ch * 16,
                      K + (size_t)(kt + key) * DIM + hbase + ch * 8);
    }
  };
  float4 rv[4];
  auto loadV = [&](int kt) {
#pragma unroll
    for (int it = 0; it < 4; ++it) {
      int i = t + it * 128;
      int key = i >> 4, c4 = (i & 15) << 2;
      rv[it] = *(const float4*)(V + (size_t)(kt + key) * DIM + hbase + c4);
    }
  };
  auto storeV = [&](int buf) {
    __bf16* dv = sVt[buf];
#pragma unroll
    for (int it = 0; it < 4; ++it) {
      int i = t + it * 128;
      int key = i >> 4, c4 = (i & 15) << 2;
      dv[(c4 + 0) * 32 + key] = f2bf(rv[it].x);
      dv[(c4 + 1) * 32 + key] = f2bf(rv[it].y);
      dv[(c4 + 2) * 32 + key] = f2bf(rv[it].z);
      dv[(c4 + 3) * 32 + key] = f2bf(rv[it].w);
    }
  };

  asyncK(0, 0);
  loadV(0);
  storeV(0);
  wait_async0();
  __syncthreads();

  FragBF aq0 = load_a(sQ + (w * 16) * 64 + 0, 64);
  FragBF aq1 = load_a(sQ + (w * 16) * 64 + 32, 64);

  float m_run[8], l_run[8];
  v8f o[4];
#pragma unroll
  for (int r = 0; r < 8; ++r) { m_run[r] = -1e30f; l_run[r] = 0.f; }
#pragma unroll
  for (int d4 = 0; d4 < 4; ++d4)
#pragma unroll
    for (int r = 0; r < 8; ++r) o[d4][r] = 0.f;

  for (int kt = 0; kt < Sk; kt += 32) {
    const int cur = (kt >> 5) & 1;
    const bool hn = (kt + 32) < Sk;
    if (hn) { asyncK(kt + 32, cur ^ 1); loadV(kt + 32); }  // overlap with compute

    const __bf16* cK = sK[cur];
    v8f s0, s1;
#pragma unroll
    for (int r = 0; r < 8; ++r) { s0[r] = 0.f; s1[r] = 0.f; }
    {
      FragBF b00 = load_b(cK + 0 * 64 + 0, 64);
      FragBF b01 = load_b(cK + 0 * 64 + 32, 64);
      FragBF b10 = load_b(cK + 16 * 64 + 0, 64);
      FragBF b11 = load_b(cK + 16 * 64 + 32, 64);
      s0 = wmma_bf16(aq0, b00, s0); s0 = wmma_bf16(aq1, b01, s0);
      s1 = wmma_bf16(aq0, b10, s1); s1 = wmma_bf16(aq1, b11, s1);
    }

    float alpha[8];
#pragma unroll
    for (int r = 0; r < 8; ++r) {
      float v0 = s0[r] * scale, v1 = s1[r] * scale;
      float mt = fmaxf(v0, v1);
#pragma unroll
      for (int off = 1; off < 16; off <<= 1)
        mt = fmaxf(mt, __shfl_xor(mt, off, 32));
      float mnew = fmaxf(m_run[r], mt);
      float a  = __expf(m_run[r] - mnew);
      float p0 = __expf(v0 - mnew);
      float p1 = __expf(v1 - mnew);
      float ls = p0 + p1;
#pragma unroll
      for (int off = 1; off < 16; off <<= 1)
        ls += __shfl_xor(ls, off, 32);
      l_run[r] = l_run[r] * a + ls;
      m_run[r] = mnew;
      alpha[r] = a;
      s0[r] = p0; s1[r] = p1;
    }
#pragma unroll
    for (int d4 = 0; d4 < 4; ++d4)
#pragma unroll
      for (int r = 0; r < 8; ++r) o[d4][r] *= alpha[r];

    // P: D-layout -> A-layout via wave-private LDS (in-order DS within wave)
    __bf16* p = sP[w];
#pragma unroll
    for (int r = 0; r < 8; ++r) {
      int pr = r + 8 * (lane >> 4);
      int pc = lane & 15;
      p[pr * 32 + pc]      = f2bf(s0[r]);
      p[pr * 32 + 16 + pc] = f2bf(s1[r]);
    }
    FragBF ap = load_a(p, 32);
#pragma unroll
    for (int d4 = 0; d4 < 4; ++d4) {
      FragBF bv = load_b(sVt[cur] + (d4 * 16) * 32, 32);
      o[d4] = wmma_bf16(ap, bv, o[d4]);
    }

    if (hn) { storeV(cur ^ 1); wait_async0(); }
    __syncthreads();
  }

#pragma unroll
  for (int d4 = 0; d4 < 4; ++d4)
#pragma unroll
    for (int r = 0; r < 8; ++r) {
      int m = q0 + w * 16 + r + 8 * (lane >> 4);
      int n = hbase + d4 * 16 + (lane & 15);
      O[(size_t)m * DIM + n] = o[d4][r] / l_run[r];
    }
}

// RoPE in-place on bf16 buffer
__global__ void rope_bf16_kernel(__bf16* __restrict__ buf,
                                 const float* __restrict__ pe, int S)
{
  int p = blockIdx.x * blockDim.x + threadIdx.x;
  if (p >= S * (DIM / 2)) return;
  int s = p / (DIM / 2), pi = p % (DIM / 2);
  int h = pi >> 5, f = pi & 31;
  int c0 = h * 64 + 2 * f;
  const float* pes = pe + (size_t)s * 128 + f * 4;
  float cosv = pes[0], sinv = pes[2];
  unsigned* pp = (unsigned*)(buf + (size_t)s * DIM + c0);
  unsigned u = *pp;
  float x0 = bf16_lo(u), x1 = bf16_hi(u);
  *pp = pack_bf16(cosv * x0 - sinv * x1, sinv * x0 + cosv * x1);
}

// RoPE f32 src -> bf16 dst (for img_to_q_control)
__global__ void rope_f32_to_bf16_kernel(const float* __restrict__ src,
                                        __bf16* __restrict__ dst,
                                        const float* __restrict__ pe, int S)
{
  int p = blockIdx.x * blockDim.x + threadIdx.x;
  if (p >= S * (DIM / 2)) return;
  int s = p / (DIM / 2), pi = p % (DIM / 2);
  int h = pi >> 5, f = pi & 31;
  int c0 = h * 64 + 2 * f;
  const float* pes = pe + (size_t)s * 128 + f * 4;
  float cosv = pes[0], sinv = pes[2];
  float x0 = src[(size_t)s * DIM + c0];
  float x1 = src[(size_t)s * DIM + c0 + 1];
  *(unsigned*)(dst + (size_t)s * DIM + c0) =
      pack_bf16(cosv * x0 - sinv * x1, sinv * x0 + cosv * x1);
}

__global__ void add_kernel(const float* __restrict__ a, const float* __restrict__ b,
                           float* __restrict__ c, int n)
{
  int i = blockIdx.x * blockDim.x + threadIdx.x;
  if (i < n) c[i] = a[i] + b[i];  // SCALE == 1.0
}

extern "C" void kernel_launch(void* const* d_in, const int* in_sizes, int n_in,
                              void* d_out, int out_size, void* d_ws, size_t ws_size,
                              hipStream_t stream)
{
  (void)in_sizes; (void)n_in; (void)out_size; (void)ws_size;
  const float* hs   = (const float*)d_in[0];
  const float* ehs  = (const float*)d_in[1];
  const float* hsc  = (const float*)d_in[2];
  const float* ehsc = (const float*)d_in[3];
  const float* i2q  = (const float*)d_in[4];
  const float* pe   = (const float*)d_in[5];
  // 0 a_q 1 a_k 2 a_v 3 a_addq 4 a_addk 5 a_addv 6 a_out 7 a_addout
  // 8 c_q 9 c_k 10 c_v 11 c_addq 12 c_addk 13 c_addv 14 c_out 15 c_addout
  const float *W[16], *Bv[16];
  for (int i = 0; i < 16; ++i) {
    W[i]  = (const float*)d_in[6 + 2 * i];
    Bv[i] = (const float*)d_in[7 + 2 * i];
  }

  const size_t SD = (size_t)SEQ * DIM;
  const size_t ID = (size_t)IMG * DIM;
  char* wsp = (char*)d_ws;
  auto alloc = [&](size_t bytes) { char* r = wsp; wsp += (bytes + 255) & ~(size_t)255; return r; };
  __bf16* q_a = (__bf16*)alloc(SD * 2);
  __bf16* k_a = (__bf16*)alloc(SD * 2);
  float*  v_a = (float*)alloc(SD * 4);
  float*  o_a = (float*)alloc(SD * 4);
  __bf16* q_c = (__bf16*)alloc(SD * 2);
  __bf16* k_c = (__bf16*)alloc(SD * 2);
  float*  v_c = (float*)alloc(SD * 4);
  float*  o_c = (float*)alloc(SD * 4);
  __bf16* q_x = (__bf16*)alloc(ID * 2);
  float*  o_x = (float*)alloc(ID * 4);
  float*  comb = v_a;  // v_a is dead after flash #1; reuse for hs + cross

  dim3 blk(256);
  dim3 gImg(DIM / 64, IMG / 128);
  dim3 gTxt(DIM / 64, TXT / 128);

  // ---- stream A ----
  gemm_bias_bf16_kernel<<<gImg, blk, 0, stream>>>(hs,  W[0], Bv[0], q_a,      IMG);
  gemm_bias_bf16_kernel<<<gTxt, blk, 0, stream>>>(ehs, W[3], Bv[3], q_a + ID, TXT);
  gemm_bias_bf16_kernel<<<gImg, blk, 0, stream>>>(hs,  W[1], Bv[1], k_a,      IMG);
  gemm_bias_bf16_kernel<<<gTxt, blk, 0, stream>>>(ehs, W[4], Bv[4], k_a + ID, TXT);
  gemm_bias_f32_kernel <<<gImg, blk, 0, stream>>>(hs,  W[2], Bv[2], v_a,      IMG);
  gemm_bias_f32_kernel <<<gTxt, blk, 0, stream>>>(ehs, W[5], Bv[5], v_a + ID, TXT);
  flash_attn_kernel<<<dim3(SEQ / 64, NHEAD), dim3(128), 0, stream>>>(q_a, k_a, v_a, o_a, SEQ);

  // ---- control stream ----
  gemm_bias_bf16_kernel<<<gImg, blk, 0, stream>>>(hsc,  W[8],  Bv[8],  q_c,      IMG);
  gemm_bias_bf16_kernel<<<gTxt, blk, 0, stream>>>(ehsc, W[11], Bv[11], q_c + ID, TXT);
  gemm_bias_bf16_kernel<<<gImg, blk, 0, stream>>>(hsc,  W[9],  Bv[9],  k_c,      IMG);
  gemm_bias_bf16_kernel<<<gTxt, blk, 0, stream>>>(ehsc, W[12], Bv[12], k_c + ID, TXT);
  gemm_bias_f32_kernel <<<gImg, blk, 0, stream>>>(hsc,  W[10], Bv[10], v_c,      IMG);
  gemm_bias_f32_kernel <<<gTxt, blk, 0, stream>>>(ehsc, W[13], Bv[13], v_c + ID, TXT);
  int rp = SEQ * (DIM / 2);
  rope_bf16_kernel<<<(rp + 255) / 256, 256, 0, stream>>>(q_c, pe, SEQ);
  rope_bf16_kernel<<<(rp + 255) / 256, 256, 0, stream>>>(k_c, pe, SEQ);  // roped kc == k_cross
  flash_attn_kernel<<<dim3(SEQ / 64, NHEAD), dim3(128), 0, stream>>>(q_c, k_c, v_c, o_c, SEQ);

  // ---- cross attention ----
  int rpx = IMG * (DIM / 2);
  rope_f32_to_bf16_kernel<<<(rpx + 255) / 256, 256, 0, stream>>>(i2q, q_x, pe, IMG);
  flash_attn_kernel<<<dim3(IMG / 64, NHEAD), dim3(128), 0, stream>>>(q_x, k_c, v_c, o_x, SEQ);

  // ---- combine + output projections ----
  add_kernel<<<((int)ID + 255) / 256, 256, 0, stream>>>(o_a, o_x, comb, (int)ID);
  float* out = (float*)d_out;
  gemm_bias_f32_kernel<<<gImg, blk, 0, stream>>>(comb,     W[6],  Bv[6],  out,                              IMG);
  gemm_bias_f32_kernel<<<gTxt, blk, 0, stream>>>(o_a + ID, W[7],  Bv[7],  out + ID,                         TXT);
  gemm_bias_f32_kernel<<<gImg, blk, 0, stream>>>(o_c,      W[14], Bv[14], out + ID + (size_t)TXT * DIM,     IMG);
  gemm_bias_f32_kernel<<<gTxt, blk, 0, stream>>>(o_c + ID, W[15], Bv[15], out + 2 * ID + (size_t)TXT * DIM, TXT);
}
